// DecisionTransformer_37211596652895
// MI455X (gfx1250) — matrix operand order repaired and checked by reading, not compiled, not run on hardware
//
#include <hip/hip_runtime.h>

#define B_ 16
#define K_ 256
#define E_ 768
#define F_ 3072
#define H_ 12
#define L_ 6
#define S_ 768            // 3*K
#define D_ 64
#define STATE_ 128
#define ACT_ 18
#define M_ (B_*S_)        // 12288 rows of activations

typedef __bf16 bf16_t;
typedef __attribute__((ext_vector_type(8)))  __bf16 v8bf;
typedef __attribute__((ext_vector_type(16))) __bf16 v16bf;
typedef __attribute__((ext_vector_type(8)))  float  v8f;

static __device__ __forceinline__ v16bf cat16(v8bf lo, v8bf hi) {
  return __builtin_shufflevector(lo, hi, 0,1,2,3,4,5,6,7,8,9,10,11,12,13,14,15);
}
// A-operand (16x32 bf16): lane l -> row l&15; contiguous 16 K-values at K = 16*h2
static __device__ __forceinline__ v16bf load_a_frag(const bf16_t* row, int k0, int h2) {
  v8bf lo = *reinterpret_cast<const v8bf*>(row + k0 + 16*h2);
  v8bf hi = *reinterpret_cast<const v8bf*>(row + k0 + 16*h2 + 8);
  return cat16(lo, hi);
}
// B-operand (32x16 bf16): lane l -> col l&15; contiguous 16 K-values at K = 16*h2
static __device__ __forceinline__ v16bf load_b_frag(const bf16_t* colrow, int k0, int h2) {
  v8bf lo = *reinterpret_cast<const v8bf*>(colrow + k0 + 16*h2);
  v8bf hi = *reinterpret_cast<const v8bf*>(colrow + k0 + 16*h2 + 8);
  return cat16(lo, hi);
}
static __device__ __forceinline__ v8f wmma_bf16(v16bf a, v16bf b, v8f c) {
  return __builtin_amdgcn_wmma_f32_16x16x32_bf16(false, a, false, b, (short)0, c, false, false);
}
static __device__ __forceinline__ v8f zero8() {
  v8f z = {0.f,0.f,0.f,0.f,0.f,0.f,0.f,0.f};
  return z;
}

// ---------------------------------------------------------------------------
// Embedding: x[b, 0:K]=R-emb, [K:2K]=s-emb, [2K:3K]=a-emb ; writes f32 + bf16
// ---------------------------------------------------------------------------
__global__ void embed_kernel(const float* __restrict__ rtg, const float* __restrict__ state,
                             const float* __restrict__ action, const float* __restrict__ timestep,
                             const float* __restrict__ Rw, const float* __restrict__ Sw,
                             const float* __restrict__ AwE, const float* __restrict__ Tw,
                             float* __restrict__ x, bf16_t* __restrict__ xb)
{
  __shared__ float ssh[STATE_];
  int bk = blockIdx.x;
  int b = bk / K_, k = bk % K_;
  int t = threadIdx.x;
  if (t < STATE_) ssh[t] = state[(size_t)bk*STATE_ + t];
  __syncthreads();
  float r = rtg[bk], a = action[bk], ts = timestep[bk];
  float tt = ts / (float)E_;
  float sgn = (tt > 0.f) ? 1.f : ((tt < 0.f) ? -1.f : 0.f);
  size_t rowR = (size_t)b*S_ + k;
  size_t rowS = rowR + K_;
  size_t rowA = rowR + 2*K_;
  for (int e = t; e < E_; e += blockDim.x) {
    float pos = sgn * Tw[e];
    float xr = r * Rw[e] + pos;
    float dot = 0.f;
    #pragma unroll 8
    for (int c = 0; c < STATE_; ++c) dot += ssh[c] * Sw[c*E_ + e];
    float xs = dot + pos;
    float xa = a * AwE[e] + pos;
    x[rowR*E_ + e] = xr;  xb[rowR*E_ + e] = (bf16_t)xr;
    x[rowS*E_ + e] = xs;  xb[rowS*E_ + e] = (bf16_t)xs;
    x[rowA*E_ + e] = xa;  xb[rowA*E_ + e] = (bf16_t)xa;
  }
}

// ---------------------------------------------------------------------------
// Weight transpose + bf16 convert: W[Kd,N] f32 -> Wt[N,Kd] bf16
// ---------------------------------------------------------------------------
__global__ void wt_kernel(const float* __restrict__ W, bf16_t* __restrict__ Wt, int Kd, int N)
{
  size_t idx = (size_t)blockIdx.x*blockDim.x + threadIdx.x;
  if (idx >= (size_t)Kd*N) return;
  int kk = (int)(idx % Kd);
  int n  = (int)(idx / Kd);
  Wt[idx] = (bf16_t)W[(size_t)kk*N + n];
}

// ---------------------------------------------------------------------------
// V transpose per head: V[B,S,E] bf16 -> Vt[B,H,D,S] bf16
// ---------------------------------------------------------------------------
__global__ void vt_kernel(const bf16_t* __restrict__ V, bf16_t* __restrict__ Vt)
{
  size_t idx = (size_t)blockIdx.x*blockDim.x + threadIdx.x; // B*H*D*S total
  int s = (int)(idx % S_);
  int d = (int)((idx / S_) % D_);
  int h = (int)((idx / ((size_t)S_*D_)) % H_);
  int b = (int)(idx / ((size_t)S_*D_*H_));
  Vt[idx] = V[((size_t)b*S_ + s)*E_ + h*D_ + d];
}

// ---------------------------------------------------------------------------
// WMMA GEMM: out[M,N] = A[M,Kd](bf16) @ Wt[N,Kd]^T + bias (+relu) (+residual)
// Wave -> 32x64 tile (2 A-frags x 4 B-frags = 8 WMMA / k-step, 12 b128 loads);
// block = 8 waves = 256 rows sharing the 64-column weight panel via L0/L2.
// ---------------------------------------------------------------------------
__global__ void gemm_kernel(const bf16_t* __restrict__ A, const bf16_t* __restrict__ Wt,
                            const float* __restrict__ bias, const float* __restrict__ residual,
                            float* __restrict__ outF, bf16_t* __restrict__ outB,
                            int M, int N, int Kd, int relu)
{
  int lane = threadIdx.x & 31, wave = threadIdx.x >> 5;
  int n_c = lane & 15, h2 = lane >> 4;
  int mBase = (blockIdx.x*8 + wave)*32;
  int nBase = blockIdx.y*64;
  const bf16_t* arow0 = A + (size_t)(mBase +  0 + n_c)*Kd;
  const bf16_t* arow1 = A + (size_t)(mBase + 16 + n_c)*Kd;
  const bf16_t* brow0 = Wt + (size_t)(nBase +  0 + n_c)*Kd;
  const bf16_t* brow1 = Wt + (size_t)(nBase + 16 + n_c)*Kd;
  const bf16_t* brow2 = Wt + (size_t)(nBase + 32 + n_c)*Kd;
  const bf16_t* brow3 = Wt + (size_t)(nBase + 48 + n_c)*Kd;
  v8f acc0[4] = {zero8(), zero8(), zero8(), zero8()};
  v8f acc1[4] = {zero8(), zero8(), zero8(), zero8()};
  for (int kk = 0; kk < Kd; kk += 32) {
    v16bf af0 = load_a_frag(arow0, kk, h2);
    v16bf af1 = load_a_frag(arow1, kk, h2);
    v16bf bf0 = load_b_frag(brow0, kk, h2);
    v16bf bf1 = load_b_frag(brow1, kk, h2);
    v16bf bf2 = load_b_frag(brow2, kk, h2);
    v16bf bf3 = load_b_frag(brow3, kk, h2);
    acc0[0] = wmma_bf16(af0, bf0, acc0[0]);
    acc0[1] = wmma_bf16(af0, bf1, acc0[1]);
    acc0[2] = wmma_bf16(af0, bf2, acc0[2]);
    acc0[3] = wmma_bf16(af0, bf3, acc0[3]);
    acc1[0] = wmma_bf16(af1, bf0, acc1[0]);
    acc1[1] = wmma_bf16(af1, bf1, acc1[1]);
    acc1[2] = wmma_bf16(af1, bf2, acc1[2]);
    acc1[3] = wmma_bf16(af1, bf3, acc1[3]);
  }
  #pragma unroll
  for (int r = 0; r < 2; ++r) {
    const v8f* acc = r ? acc1 : acc0;
    #pragma unroll
    for (int nf = 0; nf < 4; ++nf) {
      int col = nBase + nf*16 + n_c;
      float bc = bias ? bias[col] : 0.f;
      #pragma unroll
      for (int i = 0; i < 8; ++i) {
        int row = mBase + r*16 + i + 8*h2;
        float v = acc[nf][i] + bc;
        if (relu) v = v > 0.f ? v : 0.f;
        if (residual) v += residual[(size_t)row*N + col];
        if (outF) outF[(size_t)row*N + col] = v;
        if (outB) outB[(size_t)row*N + col] = (bf16_t)v;
      }
    }
  }
}

// ---------------------------------------------------------------------------
// Flash attention (causal, unscaled): one wave per (b, h, 16-query tile).
// kv step = 32 keys: scores via 4 WMMAs, P repacked via LDS, P@V via 4 WMMAs.
// ---------------------------------------------------------------------------
__global__ void attn_kernel(const bf16_t* __restrict__ Q, const bf16_t* __restrict__ Kmat,
                            const bf16_t* __restrict__ Vt, bf16_t* __restrict__ AO)
{
  __shared__ __align__(16) bf16_t Pl[16][32];
  int lane = threadIdx.x & 31;
  int n_c = lane & 15, h2 = lane >> 4;
  int qt = blockIdx.x % (S_/16);
  int h  = (blockIdx.x / (S_/16)) % H_;
  int b  = blockIdx.x / ((S_/16)*H_);
  int qBase = qt*16;

  const bf16_t* qrow = Q + ((size_t)b*S_ + qBase + n_c)*E_ + h*D_;
  v16bf qa0 = load_a_frag(qrow, 0,  h2);
  v16bf qa1 = load_a_frag(qrow, 32, h2);
  const bf16_t* kbase  = Kmat + (size_t)b*S_*E_ + h*D_;
  const bf16_t* vtbase = Vt + ((size_t)b*H_ + h)*(size_t)D_*S_;

  float mrow[8], lrow[8];
  v8f o[4] = {zero8(), zero8(), zero8(), zero8()};
  #pragma unroll
  for (int i = 0; i < 8; ++i) { mrow[i] = -3.0e38f; lrow[i] = 0.f; }

  int nsteps = (qBase + 15)/32 + 1;
  for (int j = 0; j < nsteps; ++j) {
    int jBase = j*32;
    const bf16_t* kr0 = kbase + (size_t)(jBase + n_c)*E_;
    const bf16_t* kr1 = kbase + (size_t)(jBase + 16 + n_c)*E_;
    v8f s0 = zero8(), s1 = zero8();
    s0 = wmma_bf16(qa0, load_b_frag(kr0, 0,  h2), s0);
    s0 = wmma_bf16(qa1, load_b_frag(kr0, 32, h2), s0);
    s1 = wmma_bf16(qa0, load_b_frag(kr1, 0,  h2), s1);
    s1 = wmma_bf16(qa1, load_b_frag(kr1, 32, h2), s1);

    if (jBase + 31 > qBase) {                         // diagonal tiles: causal mask
      #pragma unroll
      for (int i = 0; i < 8; ++i) {
        int qi = qBase + i + 8*h2;
        if (jBase + n_c > qi)      s0[i] = -3.0e38f;
        if (jBase + 16 + n_c > qi) s1[i] = -3.0e38f;
      }
    }
    float rmax[8], p0[8], p1[8], rsum[8], scale[8];
    #pragma unroll
    for (int i = 0; i < 8; ++i) rmax[i] = fmaxf(s0[i], s1[i]);
    #pragma unroll
    for (int off = 8; off >= 1; off >>= 1) {
      #pragma unroll
      for (int i = 0; i < 8; ++i) rmax[i] = fmaxf(rmax[i], __shfl_xor(rmax[i], off, 32));
    }
    #pragma unroll
    for (int i = 0; i < 8; ++i) {
      float mn = fmaxf(mrow[i], rmax[i]);
      scale[i] = __expf(mrow[i] - mn);
      p0[i] = __expf(s0[i] - mn);
      p1[i] = __expf(s1[i] - mn);
      rsum[i] = p0[i] + p1[i];
      mrow[i] = mn;
    }
    #pragma unroll
    for (int off = 8; off >= 1; off >>= 1) {
      #pragma unroll
      for (int i = 0; i < 8; ++i) rsum[i] += __shfl_xor(rsum[i], off, 32);
    }
    #pragma unroll
    for (int i = 0; i < 8; ++i) lrow[i] = lrow[i]*scale[i] + rsum[i];
    #pragma unroll
    for (int g = 0; g < 4; ++g)
      #pragma unroll
      for (int i = 0; i < 8; ++i) o[g][i] *= scale[i];

    // repack P from C-layout (lane=n) to A-layout (lane=m) through LDS
    __syncthreads();
    #pragma unroll
    for (int i = 0; i < 8; ++i) {
      Pl[i + 8*h2][n_c]      = (bf16_t)p0[i];
      Pl[i + 8*h2][n_c + 16] = (bf16_t)p1[i];
    }
    __syncthreads();
    v16bf pa = load_a_frag(&Pl[n_c][0], 0, h2);
    #pragma unroll
    for (int g = 0; g < 4; ++g) {
      const bf16_t* vr = vtbase + (size_t)(g*16 + n_c)*S_ + jBase;
      o[g] = wmma_bf16(pa, load_b_frag(vr, 0, h2), o[g]);
    }
    __syncthreads();
  }
  bf16_t* aorow = AO + ((size_t)b*S_ + qBase)*E_ + h*D_;
  #pragma unroll
  for (int g = 0; g < 4; ++g)
    #pragma unroll
    for (int i = 0; i < 8; ++i) {
      int row = i + 8*h2;
      aorow[(size_t)row*E_ + g*16 + n_c] = (bf16_t)(o[g][i] / lrow[i]);
    }
}

// ---------------------------------------------------------------------------
// LayerNorm over E=768: deterministic tree reduction; writes f32 + bf16
// ---------------------------------------------------------------------------
__global__ void ln_kernel(const float* __restrict__ Y, const float* __restrict__ g,
                          const float* __restrict__ be, float* __restrict__ X,
                          bf16_t* __restrict__ Xb)
{
  __shared__ float red[256], red2[256];
  int row = blockIdx.x, t = threadIdx.x;
  const float* yr = Y + (size_t)row*E_;
  float v0 = yr[t], v1 = yr[t+256], v2 = yr[t+512];
  red[t]  = v0 + v1 + v2;
  red2[t] = v0*v0 + v1*v1 + v2*v2;
  __syncthreads();
  for (int off = 128; off >= 1; off >>= 1) {
    if (t < off) { red[t] += red[t+off]; red2[t] += red2[t+off]; }
    __syncthreads();
  }
  float mean = red[0] * (1.f/E_);
  float var  = red2[0] * (1.f/E_) - mean*mean;
  float rstd = rsqrtf(var + 1e-5f);
  float* xr = X + (size_t)row*E_;
  bf16_t* xbr = Xb + (size_t)row*E_;
  {
    int e = t;       float o = (v0-mean)*rstd*g[e] + be[e]; xr[e] = o; xbr[e] = (bf16_t)o;
  }
  {
    int e = t + 256; float o = (v1-mean)*rstd*g[e] + be[e]; xr[e] = o; xbr[e] = (bf16_t)o;
  }
  {
    int e = t + 512; float o = (v2-mean)*rstd*g[e] + be[e]; xr[e] = o; xbr[e] = (bf16_t)o;
  }
}

// ---------------------------------------------------------------------------
// Head: out[b,k,a] = x[b, 2K+k] . action_w[:,a] + action_b[a]; wave per output
// ---------------------------------------------------------------------------
__global__ void head_kernel(const float* __restrict__ X, const float* __restrict__ Aw,
                            const float* __restrict__ Ab, float* __restrict__ out)
{
  int lane = threadIdx.x & 31, w = threadIdx.x >> 5;
  int gid = blockIdx.x*4 + w;                 // B*K*ACT waves total
  int a = gid % ACT_;
  int k = (gid / ACT_) % K_;
  int b = gid / (ACT_*K_);
  const float* xr = X + ((size_t)b*S_ + 2*K_ + k)*E_;
  float sum = 0.f;
  for (int e = lane; e < E_; e += 32) sum += xr[e] * Aw[(size_t)e*ACT_ + a];
  #pragma unroll
  for (int off = 16; off >= 1; off >>= 1) sum += __shfl_xor(sum, off, 32);
  if (lane == 0) out[((size_t)b*K_ + k)*ACT_ + a] = sum + Ab[a];
}

// ---------------------------------------------------------------------------
extern "C" void kernel_launch(void* const* d_in, const int* in_sizes, int n_in,
                              void* d_out, int out_size, void* d_ws, size_t ws_size,
                              hipStream_t stream)
{
  (void)in_sizes; (void)n_in; (void)out_size; (void)ws_size;
  const float* rtg      = (const float*)d_in[0];
  const float* state    = (const float*)d_in[1];
  const float* action   = (const float*)d_in[2];
  const float* timestep = (const float*)d_in[3];
  const float* Rw       = (const float*)d_in[4];
  const float* Sw       = (const float*)d_in[5];
  const float* AwE      = (const float*)d_in[6];
  const float* Tw       = (const float*)d_in[7];
  const float* qw  = (const float*)d_in[8];   const float* qbias = (const float*)d_in[9];
  const float* kw  = (const float*)d_in[10];  const float* kbias = (const float*)d_in[11];
  const float* vw  = (const float*)d_in[12];  const float* vbias = (const float*)d_in[13];
  const float* ow  = (const float*)d_in[14];  const float* obias = (const float*)d_in[15];
  const float* ln1g = (const float*)d_in[16]; const float* ln1b = (const float*)d_in[17];
  const float* f1w = (const float*)d_in[18];  const float* f1b  = (const float*)d_in[19];
  const float* f2w = (const float*)d_in[20];  const float* f2b  = (const float*)d_in[21];
  const float* ln2g = (const float*)d_in[22]; const float* ln2b = (const float*)d_in[23];
  const float* actw = (const float*)d_in[24]; const float* actb = (const float*)d_in[25];
  float* outp = (float*)d_out;

  char* ws = (char*)d_ws;
  size_t off = 0;
  auto alloc = [&](size_t bytes) -> void* {
    void* p = ws + off;
    off += (bytes + 255) & ~(size_t)255;
    return p;
  };
  float*  x    = (float*) alloc((size_t)M_*E_*4);
  float*  y    = (float*) alloc((size_t)M_*E_*4);
  bf16_t* xb   = (bf16_t*)alloc((size_t)M_*E_*2);
  bf16_t* qbuf = (bf16_t*)alloc((size_t)M_*E_*2);
  bf16_t* kbuf = (bf16_t*)alloc((size_t)M_*E_*2);
  bf16_t* vbuf = (bf16_t*)alloc((size_t)M_*E_*2);
  bf16_t* vtb  = (bf16_t*)alloc((size_t)M_*E_*2);
  bf16_t* ao   = (bf16_t*)alloc((size_t)M_*E_*2);
  bf16_t* hf   = (bf16_t*)alloc((size_t)M_*F_*2);
  bf16_t* wt   = (bf16_t*)alloc((size_t)E_*F_*2);

  embed_kernel<<<B_*K_, 256, 0, stream>>>(rtg, state, action, timestep, Rw, Sw, AwE, Tw, x, xb);

  const dim3 blk(256);
  const dim3 gE(M_/256, E_/64);   // 48 x 12  (8 waves x 32 rows per block)
  const dim3 gF(M_/256, F_/64);   // 48 x 48
  const int  gWT_EE = (E_*E_ + 255)/256;
  const int  gWT_EF = (E_*F_ + 255)/256;
  const int  gVT    = (M_*E_)/256;           // B*H*D*S elements
  const int  gAT    = B_*H_*(S_/16);         // one wave-block per q tile

  for (int l = 0; l < L_; ++l) {
    const size_t oEE = (size_t)l*E_*E_, oEF = (size_t)l*E_*F_;
    // Q, K, V projections (bf16 outputs)
    wt_kernel<<<gWT_EE, blk, 0, stream>>>(qw + oEE, wt, E_, E_);
    gemm_kernel<<<gE, blk, 0, stream>>>(xb, wt, qbias + l*E_, nullptr, nullptr, qbuf, M_, E_, E_, 0);
    wt_kernel<<<gWT_EE, blk, 0, stream>>>(kw + oEE, wt, E_, E_);
    gemm_kernel<<<gE, blk, 0, stream>>>(xb, wt, kbias + l*E_, nullptr, nullptr, kbuf, M_, E_, E_, 0);
    wt_kernel<<<gWT_EE, blk, 0, stream>>>(vw + oEE, wt, E_, E_);
    gemm_kernel<<<gE, blk, 0, stream>>>(xb, wt, vbias + l*E_, nullptr, nullptr, vbuf, M_, E_, E_, 0);
    vt_kernel<<<gVT, blk, 0, stream>>>(vbuf, vtb);
    // attention
    attn_kernel<<<gAT, 32, 0, stream>>>(qbuf, kbuf, vtb, ao);
    // output projection + residual, then LN1
    wt_kernel<<<gWT_EE, blk, 0, stream>>>(ow + oEE, wt, E_, E_);
    gemm_kernel<<<gE, blk, 0, stream>>>(ao, wt, obias + l*E_, x, y, nullptr, M_, E_, E_, 0);
    ln_kernel<<<M_, 256, 0, stream>>>(y, ln1g + l*E_, ln1b + l*E_, x, xb);
    // FFN
    wt_kernel<<<gWT_EF, blk, 0, stream>>>(f1w + oEF, wt, E_, F_);
    gemm_kernel<<<gF, blk, 0, stream>>>(xb, wt, f1b + l*F_, nullptr, nullptr, hf, M_, F_, E_, 1);
    wt_kernel<<<gWT_EF, blk, 0, stream>>>(f2w + oEF, wt, F_, E_);
    gemm_kernel<<<gE, blk, 0, stream>>>(hf, wt, f2b + l*E_, x, y, nullptr, M_, E_, F_, 0);
    ln_kernel<<<M_, 256, 0, stream>>>(y, ln2g + l*E_, ln2b + l*E_, x, xb);
  }
  head_kernel<<<(B_*K_*ACT_)/4, 128, 0, stream>>>(x, actw, actb, outp);
}
